// ProteinTransformer_46316927320099
// MI455X (gfx1250) — matrix operand order repaired and checked by reading, not compiled, not run on hardware
//
#include <hip/hip_runtime.h>

// ---------------- problem constants ----------------
#define Nn   8192
#define Kk   15
#define Ss   128
#define Ee   24
#define Hh   8
#define Aa   32
#define HIDd 512
#define DEPTH 3
#define CH    1024            // node chunk for KV/attention path
#define KVROWS (CH*Kk)        // 15360 rows per chunk
#define KVK   160             // 152 padded to multiple of 32
#define KVN   512             // K|V fused output width
#define HEADN 128             // 103 head cols padded to 128

typedef unsigned short u16;
typedef __attribute__((ext_vector_type(16))) __bf16 v16bf;
typedef __attribute__((ext_vector_type(8)))  float  v8f;

union FragAB { unsigned int u[8]; v16bf v; };

// ---------------- bf16 helpers ----------------
__device__ __forceinline__ u16 f2bf(float x) {
    unsigned int u = __float_as_uint(x);
    unsigned int r = (u + 0x7fffu + ((u >> 16) & 1u)) >> 16;   // RNE
    return (u16)r;
}
__device__ __forceinline__ float bf2f(u16 h) {
    return __uint_as_float(((unsigned int)h) << 16);
}

// ---------------- small vec3 helpers ----------------
struct V3 { float x, y, z; };
__device__ __forceinline__ V3 vsub(V3 a, V3 b){ return {a.x-b.x, a.y-b.y, a.z-b.z}; }
__device__ __forceinline__ V3 vnorm(V3 a){
    float n = sqrtf(a.x*a.x + a.y*a.y + a.z*a.z);
    float inv = 1.0f / (n + 1e-8f);
    return {a.x*inv, a.y*inv, a.z*inv};
}
__device__ __forceinline__ V3 vcross(V3 a, V3 b){
    return {a.y*b.z - a.z*b.y, a.z*b.x - a.x*b.z, a.x*b.y - a.y*b.x};
}
__device__ __forceinline__ float sgnf(float v){ return (v > 0.f) ? 1.f : ((v < 0.f) ? -1.f : 0.f); }
__device__ __forceinline__ float quat_comp(float t, float a, float b){
    return 0.5f * sqrtf(fmaxf(1e-8f, t)) * sgnf(a - b);
}

// =====================================================================
// WMMA GEMM:  C[M x N] = act(A_bf16[M x Kd] * B_bf16[Kd x N] + bias)
//   block tile 256x64 (8 waves x 32 rows), wave tile 32x64
//   double-buffered LDS, scalar register staging (no arrays -> no scratch),
//   hoisted per-thread addressing
// =====================================================================
__global__ __launch_bounds__(256, 1)   // min 1 wave/EU: allow max VGPRs, no spill
void gemm_bf16_kernel(
    const u16* __restrict__ A, int lda,
    const u16* __restrict__ B, int ldb,
    const float* __restrict__ bias,
    float* __restrict__ outF, u16* __restrict__ outB,
    int ldc, int Kd, int act, int inplaceAdd)
{
    __shared__ unsigned int Alds[2][256 * 16];  // 256 rows x 32 bf16 per buffer
    __shared__ unsigned int Blds[2][64 * 16];   // 64 cols x 32 bf16, column-major

    const int m0   = blockIdx.x * 256;
    const int n0   = blockIdx.y * 64;
    const int tid  = threadIdx.x;
    const int lane = tid & 31;
    const int wave = tid >> 5;

    v8f zero = {0.f,0.f,0.f,0.f,0.f,0.f,0.f,0.f};
    v8f acc[2][4];
    #pragma unroll
    for (int mi = 0; mi < 2; ++mi)
        #pragma unroll
        for (int nt = 0; nt < 4; ++nt) acc[mi][nt] = zero;

    // ---- per-thread invariant addressing (hoisted out of k-loop) ----
    const int rowA = tid >> 2;                       // A row within block tile
    const u16* aBase = A + (size_t)(m0 + rowA) * lda + (tid & 3) * 8;
    const size_t aStep = (size_t)64 * lda;           // +64 rows per chunk
    const int kB = tid >> 4;                         // B row (k) within tile
    const u16* bBase = B + (size_t)kB * ldb + n0 + (tid & 15) * 4;
    const size_t bStep = (size_t)16 * ldb;           // +16 k per chunk
    unsigned int* aldsBase0 = &Alds[0][rowA * 16 + (tid & 3) * 4];
    unsigned int* aldsBase1 = &Alds[1][rowA * 16 + (tid & 3) * 4];
    u16* bldsBase0 = (u16*)Blds[0] + (tid & 15) * 128 + kB;
    u16* bldsBase1 = (u16*)Blds[1] + (tid & 15) * 128 + kB;

    // scalar staging registers (SROA-friendly: no arrays, no lambdas)
    uint4 a0, a1, a2, a3;
    uint2 b0, b1;

#define LOAD_TILE(kt)                                                       \
    { const u16* ap = aBase + (size_t)(kt) * 32;                            \
      a0 = *(const uint4*)(ap);                                             \
      a1 = *(const uint4*)(ap + aStep);                                     \
      a2 = *(const uint4*)(ap + 2 * aStep);                                 \
      a3 = *(const uint4*)(ap + 3 * aStep);                                 \
      const u16* bp = bBase + (size_t)(kt) * 32 * ldb;                      \
      b0 = *(const uint2*)(bp);                                             \
      b1 = *(const uint2*)(bp + bStep); }

#define STORE_TILE(al, bl)                                                  \
    { *(uint4*)((al))          = a0;                                        \
      *(uint4*)((al) + 1024)   = a1;                                        \
      *(uint4*)((al) + 2048)   = a2;                                        \
      *(uint4*)((al) + 3072)   = a3;                                        \
      (bl)[0]   = (u16)(b0.x & 0xffffu);                                    \
      (bl)[32]  = (u16)(b0.x >> 16);                                        \
      (bl)[64]  = (u16)(b0.y & 0xffffu);                                    \
      (bl)[96]  = (u16)(b0.y >> 16);                                        \
      (bl)[16]  = (u16)(b1.x & 0xffffu);                                    \
      (bl)[48]  = (u16)(b1.x >> 16);                                        \
      (bl)[80]  = (u16)(b1.y & 0xffffu);                                    \
      (bl)[112] = (u16)(b1.y >> 16); }

    const int nK = Kd >> 5;
    int buf = 0;
    LOAD_TILE(0);
    STORE_TILE(aldsBase0, bldsBase0);
    __syncthreads();

    for (int kt = 0; kt < nK; ++kt) {
        bool hasNext = (kt + 1) < nK;
        if (hasNext) LOAD_TILE(kt + 1);            // overlaps WMMA below

        // ---- fragments (documented 16-bit 16x32 A / 32x16 B layouts) ----
        FragAB af[2];
        {
            int kb = (lane < 16) ? 0 : 8;
            #pragma unroll
            for (int mi = 0; mi < 2; ++mi) {
                int mrow = wave * 32 + mi * 16 + (lane & 15);
                #pragma unroll
                for (int v = 0; v < 8; ++v) {
                    int kp = (v < 4) ? (kb + 2 * v) : (16 + kb + 2 * (v - 4));
                    af[mi].u[v] = Alds[buf][mrow * 16 + (kp >> 1)];
                }
            }
        }
        int kbB = (lane < 16) ? 0 : 8;
        #pragma unroll
        for (int nt = 0; nt < 4; ++nt) {
            FragAB bfr;
            int ncol = nt * 16 + (lane & 15);
            #pragma unroll
            for (int v = 0; v < 8; ++v)
                bfr.u[v] = Blds[buf][ncol * 16 + (((v < 4) ? (kbB + 2 * v) : (16 + kbB + 2 * (v - 4))) >> 1)];
            acc[0][nt] = __builtin_amdgcn_wmma_f32_16x16x32_bf16(
                false, af[0].v, false, bfr.v, (short)0, acc[0][nt], false, false);
            acc[1][nt] = __builtin_amdgcn_wmma_f32_16x16x32_bf16(
                false, af[1].v, false, bfr.v, (short)0, acc[1][nt], false, false);
        }

        if (hasNext) {
            if (buf == 0) { STORE_TILE(aldsBase1, bldsBase1); }
            else          { STORE_TILE(aldsBase0, bldsBase0); }
            __syncthreads();
            buf ^= 1;
        }
    }
#undef LOAD_TILE
#undef STORE_TILE

    // ---- epilogue: documented C layout (vgpr r: M=r(+8), N=lane%16) ----
    const int mloc = (lane >= 16) ? 8 : 0;
    #pragma unroll
    for (int mi = 0; mi < 2; ++mi) {
        #pragma unroll
        for (int nt = 0; nt < 4; ++nt) {
            int n = n0 + nt * 16 + (lane & 15);
            float bvv = bias ? bias[n] : 0.f;
            #pragma unroll
            for (int r = 0; r < 8; ++r) {
                int m = m0 + wave * 32 + mi * 16 + mloc + r;
                float val = acc[mi][nt][r] + bvv;
                if (act) val = fmaxf(val, 0.f);
                size_t idx = (size_t)m * ldc + n;
                if (outF) {
                    if (inplaceAdd) outF[idx] += val;
                    else            outF[idx]  = val;
                }
                if (outB) outB[idx] = f2bf(val);
            }
        }
    }
}

// =====================================================================
// weight packing / conversion
// =====================================================================
__global__ void convert_kernel(const float* __restrict__ src, u16* __restrict__ dst, int n) {
    int i = blockIdx.x * blockDim.x + threadIdx.x;
    if (i < n) dst[i] = f2bf(src[i]);
}

__global__ void pack_kv_kernel(const float* __restrict__ Wk, const float* __restrict__ Wv,
                               const float* __restrict__ bk, const float* __restrict__ bv,
                               u16* __restrict__ Wkv, float* __restrict__ bkv, int d) {
    int i = blockIdx.x * blockDim.x + threadIdx.x;
    if (i >= KVK * KVN) return;
    int r = i >> 9, c = i & 511;
    float v = 0.f;
    if (r < Ss + Ee)
        v = (c < 256) ? Wk[((size_t)d * (Ss + Ee) + r) * 256 + c]
                      : Wv[((size_t)d * (Ss + Ee) + r) * 256 + (c - 256)];
    Wkv[(size_t)d * KVK * KVN + i] = f2bf(v);
    if (i < KVN) bkv[d * KVN + i] = (i < 256) ? bk[d * 256 + i] : bv[d * 256 + i - 256];
}

__global__ void pack_head_kernel(const float* __restrict__ Ws, const float* __restrict__ Wc,
                                 const float* __restrict__ Wco, const float* __restrict__ Wwt,
                                 const float* __restrict__ Wfac,
                                 const float* __restrict__ bs, const float* __restrict__ bc,
                                 const float* __restrict__ bco, const float* __restrict__ bwt,
                                 const float* __restrict__ bfac,
                                 u16* __restrict__ Wh, float* __restrict__ bh) {
    int i = blockIdx.x * blockDim.x + threadIdx.x;
    if (i >= Ss * HEADN) return;
    int r = i >> 7, c = i & 127;
    float v = 0.f;
    if      (c < 30)  v = Ws [r * 30 + c];
    else if (c < 60)  v = Wc [r * 30 + c - 30];
    else if (c < 90)  v = Wco[r * 30 + c - 60];
    else if (c < 100) v = Wwt[r * 10 + c - 90];
    else if (c < 103) v = Wfac[r * 3 + c - 100];
    Wh[i] = f2bf(v);
    if (i < HEADN) {
        float b = 0.f;
        if      (i < 30)  b = bs [i];
        else if (i < 60)  b = bc [i - 30];
        else if (i < 90)  b = bco[i - 60];
        else if (i < 100) b = bwt[i - 90];
        else if (i < 103) b = bfac[i - 100];
        bh[i] = b;
    }
}

// =====================================================================
// geometry: pos extraction + orientation frames (columns b, n, o)
// =====================================================================
__global__ void geom_kernel(const float* __restrict__ tert,
                            float* __restrict__ pos, float* __restrict__ Rm) {
    int i = blockIdx.x * blockDim.x + threadIdx.x;
    if (i >= Nn) return;
    pos[i*3+0] = tert[i*9+3]; pos[i*3+1] = tert[i*9+4]; pos[i*3+2] = tert[i*9+5];
    int j = (i < 1) ? 1 : ((i > Nn-2) ? Nn-2 : i);
    V3 p0 = {tert[(j-1)*9+3], tert[(j-1)*9+4], tert[(j-1)*9+5]};
    V3 p1 = {tert[(j  )*9+3], tert[(j  )*9+4], tert[(j  )*9+5]};
    V3 p2 = {tert[(j+1)*9+3], tert[(j+1)*9+4], tert[(j+1)*9+5]};
    V3 u0 = vnorm(vsub(p1, p0));
    V3 u1 = vnorm(vsub(p2, p1));
    V3 b  = vnorm(vsub(u0, u1));
    V3 nv = vnorm(vcross(u0, u1));
    V3 o  = vcross(b, nv);
    float* R = Rm + (size_t)i * 9;
    R[0]=b.x; R[1]=nv.x; R[2]=o.x;
    R[3]=b.y; R[4]=nv.y; R[5]=o.y;
    R[6]=b.z; R[7]=nv.z; R[8]=o.z;
}

// edge features -> bf16 [N*K x 24]: rbf(16) | direction(3) | quat(4) | rel(1)
__global__ void edges_kernel(const float* __restrict__ pos, const float* __restrict__ Rm,
                             const int* __restrict__ structure, u16* __restrict__ edges) {
    int idx = blockIdx.x * blockDim.x + threadIdx.x;
    if (idx >= Nn * Kk) return;
    int n  = idx / Kk;
    int tI = (n - 1 + Nn) % Nn;             // roll(…,1) target index
    int s  = structure[idx];
    const float* Rt = Rm + (size_t)tI * 9;
    const float* Rs = Rm + (size_t)s  * 9;
    V3 pt = {pos[tI*3], pos[tI*3+1], pos[tI*3+2]};
    V3 ps = {pos[s*3],  pos[s*3+1],  pos[s*3+2]};
    V3 dv = vsub(ps, pt);
    float dist = sqrtf(dv.x*dv.x + dv.y*dv.y + dv.z*dv.z);
    float inv  = 1.0f / (dist + 1e-8f);
    V3 dn = {dv.x*inv, dv.y*inv, dv.z*inv};
    float dir0 = Rt[0]*dn.x + Rt[3]*dn.y + Rt[6]*dn.z;
    float dir1 = Rt[1]*dn.x + Rt[4]*dn.y + Rt[7]*dn.z;
    float dir2 = Rt[2]*dn.x + Rt[5]*dn.y + Rt[8]*dn.z;
    float Rr[3][3];
    #pragma unroll
    for (int i2 = 0; i2 < 3; ++i2)
        #pragma unroll
        for (int l = 0; l < 3; ++l)
            Rr[i2][l] = Rt[0*3+i2]*Rs[0*3+l] + Rt[1*3+i2]*Rs[1*3+l] + Rt[2*3+i2]*Rs[2*3+l];
    float m00 = Rr[0][0], m11 = Rr[1][1], m22 = Rr[2][2];
    float qw = 0.5f * sqrtf(fmaxf(1e-8f, 1.f + m00 + m11 + m22));
    float qx = quat_comp(1.f + m00 - m11 - m22, Rr[2][1], Rr[1][2]);
    float qy = quat_comp(1.f - m00 + m11 - m22, Rr[0][2], Rr[2][0]);
    float qz = quat_comp(1.f - m00 - m11 + m22, Rr[1][0], Rr[0][1]);
    u16* e = edges + (size_t)idx * Ee;
    const float sig_inv = 1.0f / 1.25f;           // RBF_MAX / RBF_K
    #pragma unroll
    for (int j = 0; j < 16; ++j) {
        float mu = (20.0f / 15.0f) * (float)j;    // linspace(0,20,16)
        float t  = (dist - mu) * sig_inv;
        e[j] = f2bf(expf(-t * t));
    }
    e[16] = f2bf(dir0); e[17] = f2bf(dir1); e[18] = f2bf(dir2);
    e[19] = f2bf(qw); e[20] = f2bf(qx); e[21] = f2bf(qy); e[22] = f2bf(qz);
    e[23] = f2bf((float)(s - tI) * 0.01f);
}

// angle embedding: feats = [sin(prev),cos(prev)] @ W_in + b_in  (bf16 out)
__global__ void feats_kernel(const float* __restrict__ angles, const float* __restrict__ W_in,
                             const float* __restrict__ b_in, u16* __restrict__ feats) {
    int idx = blockIdx.x * blockDim.x + threadIdx.x;
    if (idx >= Nn * Ss) return;
    int n = idx >> 7, c = idx & 127;
    float p0 = 0.f, p1 = 0.f, p2 = 0.f;
    if (n > 0) { p0 = angles[(n-1)*3]; p1 = angles[(n-1)*3+1]; p2 = angles[(n-1)*3+2]; }
    float af[6] = {sinf(p0), sinf(p1), sinf(p2), cosf(p0), cosf(p1), cosf(p2)};
    float s = b_in[c];
    #pragma unroll
    for (int j = 0; j < 6; ++j) s += af[j] * W_in[j * Ss + c];
    feats[idx] = f2bf(s);
}

// LayerNorm (f32 in, bf16 out), one block per row
__global__ __launch_bounds__(128) void ln_kernel(const float* __restrict__ x,
                                                 const float* __restrict__ g,
                                                 const float* __restrict__ b,
                                                 u16* __restrict__ out) {
    __shared__ float buf[128];
    int n = blockIdx.x, c = threadIdx.x;
    float v = x[(size_t)n * Ss + c];
    buf[c] = v; __syncthreads();
    for (int s2 = 64; s2 > 0; s2 >>= 1) { if (c < s2) buf[c] += buf[c + s2]; __syncthreads(); }
    float mean = buf[0] * (1.0f / 128.0f);
    __syncthreads();
    float d = v - mean;
    buf[c] = d * d; __syncthreads();
    for (int s2 = 64; s2 > 0; s2 >>= 1) { if (c < s2) buf[c] += buf[c + s2]; __syncthreads(); }
    float var = buf[0] * (1.0f / 128.0f);
    out[(size_t)n * Ss + c] = f2bf(d * rsqrtf(var + 1e-5f) * g[c] + b[c]);
}

// gather kv_in rows: [nx[structure] (128) | edges (24) | zero pad (8)]
__global__ void gather_kv_kernel(const u16* __restrict__ nx, const u16* __restrict__ edges,
                                 const int* __restrict__ structure,
                                 u16* __restrict__ kv_in, int n0) {
    int i = blockIdx.x * blockDim.x + threadIdx.x;
    if (i >= KVROWS * KVK) return;
    int row = i / KVK, c = i - row * KVK;
    int gn = n0 + row / Kk;
    int gk = row % Kk;
    u16 v = 0;
    if (c < Ss) {
        int s = structure[gn * Kk + gk];
        v = nx[(size_t)s * Ss + c];
    } else if (c < Ss + Ee) {
        v = edges[((size_t)gn * Kk + gk) * Ee + (c - Ss)];
    }
    kv_in[(size_t)row * KVK + c] = v;
}

// attention: one block (128 thr) per node; K=15 neighbors, H=8 heads, A=32
__global__ __launch_bounds__(128) void attn_kernel(const float* __restrict__ q,
                                                   const u16* __restrict__ kv,
                                                   u16* __restrict__ att, int n0) {
    __shared__ float qs[256];
    __shared__ float lg[Kk * Hh];
    __shared__ float aw[Kk * Hh];
    int rn = blockIdx.x;
    int n  = n0 + rn;
    int t  = threadIdx.x;
    qs[t]       = q[(size_t)n * 256 + t];
    qs[t + 128] = q[(size_t)n * 256 + t + 128];
    __syncthreads();
    if (t < Kk * Hh) {
        int k = t / Hh, h = t % Hh;
        const u16* krow = kv + ((size_t)(rn * Kk + k)) * KVN + h * Aa;
        float s = 0.f;
        #pragma unroll
        for (int a = 0; a < Aa; ++a) s += qs[h * Aa + a] * bf2f(krow[a]);
        lg[t] = s * 0.17677669529663687f;   // 1/sqrt(32)
    }
    __syncthreads();
    if (t < Hh) {
        float mx = -1e30f;
        for (int k = 0; k < Kk; ++k) mx = fmaxf(mx, lg[k * Hh + t]);
        float sum = 0.f;
        for (int k = 0; k < Kk; ++k) { float e = expf(lg[k * Hh + t] - mx); aw[k * Hh + t] = e; sum += e; }
        float inv = 1.0f / sum;
        for (int k = 0; k < Kk; ++k) aw[k * Hh + t] *= inv;
    }
    __syncthreads();
    for (int o = t; o < 256; o += 128) {
        int h = o >> 5, a = o & 31;
        float s = 0.f;
        #pragma unroll
        for (int k = 0; k < Kk; ++k)
            s += aw[k * Hh + h] * bf2f(kv[((size_t)(rn * Kk + k)) * KVN + 256 + h * Aa + a]);
        att[(size_t)n * 256 + o] = f2bf(s);
    }
}

// final heads: atan2 / factor adjust / sigmoid-conc / softmax-weights
__global__ void head_final_kernel(const float* __restrict__ hr, const float* __restrict__ angles,
                                  float* __restrict__ out) {
    int n = blockIdx.x * blockDim.x + threadIdx.x;
    if (n >= Nn) return;
    const float* r = hr + (size_t)n * HEADN;
    float a0 = angles[n*3], a1 = angles[n*3+1];
    float f0 = r[100], f1 = r[101], f2 = r[102];
    float mx = -1e30f;
    for (int j = 0; j < 10; ++j) mx = fmaxf(mx, r[90 + j]);
    float sum = 0.f, e[10];
    for (int j = 0; j < 10; ++j) { e[j] = expf(r[90 + j] - mx); sum += e[j]; }
    float inv = 1.0f / sum;
    for (int j = 0; j < 10; ++j) out[(size_t)n * 10 + j] = e[j] * inv;
    float* meanOut = out + (size_t)Nn * 10 + (size_t)n * 30;
    float* concOut = out + (size_t)Nn * 40 + (size_t)n * 30;
    for (int j = 0; j < 30; ++j) {
        float m = atan2f(r[j], r[30 + j]);
        int row = j / 10;
        if (row == 1)      m += f0 * a0;
        else if (row == 2) m += f1 * a0 + f2 * a1;
        meanOut[j] = m;
        concOut[j] = 0.1f + 1000.0f / (1.0f + expf(-r[60 + j]));
    }
}

// =====================================================================
// host orchestration
// =====================================================================
static void launch_gemm(hipStream_t s, const u16* A, int lda, const u16* B, int ldb,
                        const float* bias, float* outF, u16* outB, int ldc,
                        int M, int Kd, int act, int inplaceAdd) {
    dim3 grid(M / 256, ldb / 64);
    gemm_bf16_kernel<<<grid, 256, 0, s>>>(A, lda, B, ldb, bias, outF, outB, ldc, Kd, act, inplaceAdd);
}

extern "C" void kernel_launch(void* const* d_in, const int* in_sizes, int n_in,
                              void* d_out, int out_size, void* d_ws, size_t ws_size,
                              hipStream_t stream) {
    (void)in_sizes; (void)n_in; (void)out_size; (void)ws_size;
    const float* angles = (const float*)d_in[0];
    const float* tert   = (const float*)d_in[1];
    const int*   structure = (const int*)d_in[2];
    // d_in[3] = subgraph (unused by reference math)
    const float* W_in = (const float*)d_in[4];  const float* b_in = (const float*)d_in[5];
    const float* W_pre= (const float*)d_in[6];  const float* b_pre= (const float*)d_in[7];
    const float* ln1_g= (const float*)d_in[8];  const float* ln1_b= (const float*)d_in[9];
    const float* ln2_g= (const float*)d_in[10]; const float* ln2_b= (const float*)d_in[11];
    const float* Wq   = (const float*)d_in[12]; const float* bq   = (const float*)d_in[13];
    const float* Wk   = (const float*)d_in[14]; const float* bk   = (const float*)d_in[15];
    const float* Wv   = (const float*)d_in[16]; const float* bv   = (const float*)d_in[17];
    const float* Wo   = (const float*)d_in[18]; const float* bo   = (const float*)d_in[19];
    const float* W1   = (const float*)d_in[20]; const float* b1   = (const float*)d_in[21];
    const float* W2   = (const float*)d_in[22]; const float* b2   = (const float*)d_in[23];
    const float* W3   = (const float*)d_in[24]; const float* b3   = (const float*)d_in[25];
    const float* Wsm  = (const float*)d_in[26]; const float* bsm  = (const float*)d_in[27];
    const float* Wcm  = (const float*)d_in[28]; const float* bcm  = (const float*)d_in[29];
    const float* Wconc= (const float*)d_in[30]; const float* bconc= (const float*)d_in[31];
    const float* Wwt  = (const float*)d_in[32]; const float* bwt  = (const float*)d_in[33];
    const float* Wfac = (const float*)d_in[34]; const float* bfac = (const float*)d_in[35];
    float* out = (float*)d_out;

    // ---- workspace carve (256B aligned) ----
    char* base = (char*)d_ws;
    size_t off = 0;
    auto carve = [&](size_t bytes) -> void* {
        void* p = base + off;
        off += (bytes + 255) & ~(size_t)255;
        return p;
    };
    float* pos     = (float*)carve((size_t)Nn * 3 * 4);
    float* Rm      = (float*)carve((size_t)Nn * 9 * 4);
    u16*   edges   = (u16*)  carve((size_t)Nn * Kk * Ee * 2);
    u16*   feats   = (u16*)  carve((size_t)Nn * Ss * 2);
    float* x       = (float*)carve((size_t)Nn * Ss * 4);
    u16*   nx      = (u16*)  carve((size_t)Nn * Ss * 2);
    float* qb      = (float*)carve((size_t)Nn * 256 * 4);
    u16*   att     = (u16*)  carve((size_t)Nn * 256 * 2);
    u16*   h1      = (u16*)  carve((size_t)Nn * HIDd * 2);
    u16*   h2      = (u16*)  carve((size_t)Nn * HIDd * 2);
    float* headraw = (float*)carve((size_t)Nn * HEADN * 4);
    u16*   kvin    = (u16*)  carve((size_t)KVROWS * KVK * 2);
    u16*   kvout   = (u16*)  carve((size_t)KVROWS * KVN * 2);
    u16*   Wpre_b  = (u16*)  carve((size_t)Ss * Ss * 2);
    u16*   Wq_b    = (u16*)  carve((size_t)DEPTH * Ss * 256 * 2);
    u16*   Wkv_b   = (u16*)  carve((size_t)DEPTH * KVK * KVN * 2);
    u16*   Wo_b    = (u16*)  carve((size_t)DEPTH * 256 * Ss * 2);
    u16*   W1_b    = (u16*)  carve((size_t)DEPTH * Ss * HIDd * 2);
    u16*   W2_b    = (u16*)  carve((size_t)DEPTH * HIDd * HIDd * 2);
    u16*   W3_b    = (u16*)  carve((size_t)DEPTH * HIDd * Ss * 2);
    u16*   Wh_b    = (u16*)  carve((size_t)Ss * HEADN * 2);
    float* bkv     = (float*)carve((size_t)DEPTH * KVN * 4);
    float* bh      = (float*)carve((size_t)HEADN * 4);

    const int TB = 256;
    auto blocks = [](int n, int tb) { return (n + tb - 1) / tb; };

    // ---- 1. weight conversion / packing ----
    convert_kernel<<<blocks(Ss*Ss, TB), TB, 0, stream>>>(W_pre, Wpre_b, Ss*Ss);
    convert_kernel<<<blocks(DEPTH*Ss*256, TB), TB, 0, stream>>>(Wq, Wq_b, DEPTH*Ss*256);
    convert_kernel<<<blocks(DEPTH*256*Ss, TB), TB, 0, stream>>>(Wo, Wo_b, DEPTH*256*Ss);
    convert_kernel<<<blocks(DEPTH*Ss*HIDd, TB), TB, 0, stream>>>(W1, W1_b, DEPTH*Ss*HIDd);
    convert_kernel<<<blocks(DEPTH*HIDd*HIDd, TB), TB, 0, stream>>>(W2, W2_b, DEPTH*HIDd*HIDd);
    convert_kernel<<<blocks(DEPTH*HIDd*Ss, TB), TB, 0, stream>>>(W3, W3_b, DEPTH*HIDd*Ss);
    for (int d = 0; d < DEPTH; ++d)
        pack_kv_kernel<<<blocks(KVK*KVN, TB), TB, 0, stream>>>(Wk, Wv, bk, bv, Wkv_b, bkv, d);
    pack_head_kernel<<<blocks(Ss*HEADN, TB), TB, 0, stream>>>(
        Wsm, Wcm, Wconc, Wwt, Wfac, bsm, bcm, bconc, bwt, bfac, Wh_b, bh);

    // ---- 2. geometry + features ----
    geom_kernel<<<blocks(Nn, TB), TB, 0, stream>>>(tert, pos, Rm);
    feats_kernel<<<blocks(Nn*Ss, TB), TB, 0, stream>>>(angles, W_in, b_in, feats);
    edges_kernel<<<blocks(Nn*Kk, TB), TB, 0, stream>>>(pos, Rm, structure, edges);

    // ---- 3. x = feats @ W_pre + b_pre ----
    launch_gemm(stream, feats, Ss, Wpre_b, Ss, b_pre, x, nullptr, Ss, Nn, Ss, 0, 0);

    // ---- 4. transformer layers ----
    for (int d = 0; d < DEPTH; ++d) {
        ln_kernel<<<Nn, 128, 0, stream>>>(x, ln1_g + d*Ss, ln1_b + d*Ss, nx);
        launch_gemm(stream, nx, Ss, Wq_b + (size_t)d*Ss*256, 256, bq + d*256,
                    qb, nullptr, 256, Nn, Ss, 0, 0);
        for (int c0 = 0; c0 < Nn; c0 += CH) {
            gather_kv_kernel<<<blocks(KVROWS*KVK, TB), TB, 0, stream>>>(nx, edges, structure, kvin, c0);
            launch_gemm(stream, kvin, KVK, Wkv_b + (size_t)d*KVK*KVN, KVN, bkv + d*KVN,
                        nullptr, kvout, KVN, KVROWS, KVK, 0, 0);
            attn_kernel<<<CH, 128, 0, stream>>>(qb, kvout, att, c0);
        }
        launch_gemm(stream, att, 256, Wo_b + (size_t)d*256*Ss, Ss, bo + d*Ss,
                    x, nullptr, Ss, Nn, 256, 0, 1);          // x += att@Wo + bo
        ln_kernel<<<Nn, 128, 0, stream>>>(x, ln2_g + d*Ss, ln2_b + d*Ss, nx);
        launch_gemm(stream, nx, Ss, W1_b + (size_t)d*Ss*HIDd, HIDd, b1 + d*HIDd,
                    nullptr, h1, HIDd, Nn, Ss, 1, 0);        // relu
        launch_gemm(stream, h1, HIDd, W2_b + (size_t)d*HIDd*HIDd, HIDd, b2 + d*HIDd,
                    nullptr, h2, HIDd, Nn, HIDd, 1, 0);      // relu
        launch_gemm(stream, h2, HIDd, W3_b + (size_t)d*HIDd*Ss, Ss, b3 + d*Ss,
                    x, nullptr, Ss, Nn, HIDd, 0, 1);         // x += h@W3 + b3
    }

    // ---- 5. heads ----
    convert_kernel<<<blocks(Nn*Ss, TB), TB, 0, stream>>>(x, nx, Nn*Ss);   // enc -> bf16
    launch_gemm(stream, nx, Ss, Wh_b, HEADN, bh, headraw, nullptr, HEADN, Nn, Ss, 0, 0);
    head_final_kernel<<<blocks(Nn, TB), TB, 0, stream>>>(headraw, angles, out);
}
